// MHA__15470472200192
// MI455X (gfx1250) — matrix operand order repaired and checked by reading, not compiled, not run on hardware
//
#include <hip/hip_runtime.h>

// ---------------------------------------------------------------------------
// Two-head linear attention, associativity-rewritten:
//   O_h = X_h * (Wq_h^T Wk_h * (X_h^T X_h) * Wv_h^T),  X_h = x[:, h*512:(h+1)*512]
// All GEMMs: bf16 operands, f32 WMMA accumulate, "TN" storage
// (lhs [M][K], rhs [N][K], K contiguous -> fragments load as contiguous b128s).
// Wave tile: 64x64 = 4x4 v_wmma_f32_16x16x32_bf16 tiles, 16 wmma : 16 b128/iter.
// ---------------------------------------------------------------------------

typedef __bf16 bf16_t;
typedef __attribute__((ext_vector_type(16))) __bf16 v16bf;
typedef __attribute__((ext_vector_type(8)))  __bf16 v8bf;
typedef __attribute__((ext_vector_type(4)))  __bf16 v4bf;
typedef __attribute__((ext_vector_type(4)))  float  v4f;
typedef __attribute__((ext_vector_type(8)))  float  v8f;

#define NTOK  4096
#define DFULL 1024
#define DHALF 512

// ---------------- conversion kernels ----------------

__global__ __launch_bounds__(256) void k_convert4(const float* __restrict__ src,
                                                  bf16_t* __restrict__ dst, int n4) {
  int i = blockIdx.x * blockDim.x + threadIdx.x;
  if (i < n4) {
    v4f v = reinterpret_cast<const v4f*>(src)[i];
    v4bf o;
    o[0] = (bf16_t)v[0]; o[1] = (bf16_t)v[1];
    o[2] = (bf16_t)v[2]; o[3] = (bf16_t)v[3];
    reinterpret_cast<v4bf*>(dst)[i] = o;
  }
}

// dst[c*rows + r] = (bf16) src[r*cols + c]   (rows, cols multiples of 32)
__global__ __launch_bounds__(256) void k_transpose_convert(const float* __restrict__ src,
                                                           bf16_t* __restrict__ dst,
                                                           int rows, int cols) {
  __shared__ float tile[32][33];
  int c0 = blockIdx.x * 32, r0 = blockIdx.y * 32;
  int tx = threadIdx.x, ty = threadIdx.y;   // blockDim = (32, 8)
#pragma unroll
  for (int i = 0; i < 32; i += 8)
    tile[ty + i][tx] = src[(size_t)(r0 + ty + i) * cols + (c0 + tx)];
  __syncthreads();
#pragma unroll
  for (int i = 0; i < 32; i += 8)
    dst[(size_t)(c0 + ty + i) * rows + (r0 + tx)] = (bf16_t)tile[tx][ty + i];
}

// ---------------- generic TN wave-tile GEMM (bf16 -> f32 acc) ----------------
// Fragment layouts per CDNA5 ISA 7.12.2 (wave32):
//  A 16x32 bf16: lane<16 holds K 0..7 (v0..v3) and 16..23 (v4..v7);
//                lane>=16 holds K 8..15 and 24..31.   row = lane%16
//  B 32x16 bf16: same K split as A (lane<16: K 0..7,16..23; lane>=16: K 8..15,24..31). col = lane%16
//  C/D 16x16 f32: VGPR r <-> row 8*half + r, col = lane%16

__device__ inline v16bf load_a_frag(const bf16_t* __restrict__ arow, int koff) {
  v8bf lo = *reinterpret_cast<const v8bf*>(arow + koff);        // K koff..koff+7
  v8bf hi = *reinterpret_cast<const v8bf*>(arow + 16 + koff);   // K 16+koff..
  return __builtin_shufflevector(lo, hi, 0, 1, 2, 3, 4, 5, 6, 7,
                                         8, 9, 10, 11, 12, 13, 14, 15);
}

__device__ inline v16bf load_b_frag(const bf16_t* __restrict__ brow, int boff) {
  v8bf lo = *reinterpret_cast<const v8bf*>(brow + boff);        // K boff..boff+7
  v8bf hi = *reinterpret_cast<const v8bf*>(brow + 16 + boff);   // K 16+boff..
  return __builtin_shufflevector(lo, hi, 0, 1, 2, 3, 4, 5, 6, 7,
                                         8, 9, 10, 11, 12, 13, 14, 15);
}

enum { STORE_BF16 = 0, STORE_BF16_T = 1, STORE_F32 = 2 };

__global__ __launch_bounds__(128) void k_gemm_tn(const bf16_t* __restrict__ A, int lda,
                                                 const bf16_t* __restrict__ B, int ldb,
                                                 void* __restrict__ Cp, int ldc,
                                                 int M, int N, int K, int mode) {
  const int lane = threadIdx.x & 31;
  const int wave = threadIdx.x >> 5;
  const int hf   = lane >> 4;                 // half-wave select
  const int lrow = lane & 15;
  const int koff = hf * 8;                    // A-frag K sub-offset
  const int boff = hf * 8;                    // B-frag K sub-offset

  const int tiles_n = N >> 6;                 // 64-wide column tiles
  const int t  = blockIdx.x * 4 + wave;       // 4 waves / block, one 64x64 tile each
  const int i0 = (t / tiles_n) * 64;
  const int j0 = (t % tiles_n) * 64;

  const bf16_t* ar[4];
  const bf16_t* br[4];
#pragma unroll
  for (int r = 0; r < 4; ++r) {
    ar[r] = A + (size_t)(i0 + r * 16 + lrow) * lda;
    br[r] = B + (size_t)(j0 + r * 16 + lrow) * ldb;
  }

  v8f acc[4][4];
#pragma unroll
  for (int r = 0; r < 4; ++r)
#pragma unroll
    for (int c = 0; c < 4; ++c) acc[r][c] = (v8f){};

  for (int k0 = 0; k0 < K; k0 += 32) {
    if (k0 + 256 < K) {                       // uniform branch; EXEC stays full
      __builtin_prefetch(ar[0] + k0 + 256, 0, 1);   // global_prefetch_b8
      __builtin_prefetch(br[0] + k0 + 256, 0, 1);
    }
    v16bf af[4], bfr[4];
#pragma unroll
    for (int r = 0; r < 4; ++r) af[r]  = load_a_frag(ar[r] + k0, koff);
#pragma unroll
    for (int c = 0; c < 4; ++c) bfr[c] = load_b_frag(br[c] + k0, boff);
#pragma unroll
    for (int r = 0; r < 4; ++r)
#pragma unroll
      for (int c = 0; c < 4; ++c)
        acc[r][c] = __builtin_amdgcn_wmma_f32_16x16x32_bf16(
            false, af[r], false, bfr[c], (short)0, acc[r][c], false, false);
  }

#pragma unroll
  for (int r = 0; r < 4; ++r) {
#pragma unroll
    for (int c = 0; c < 4; ++c) {
      const v8f a = acc[r][c];
      const int mbase = i0 + r * 16 + hf * 8;       // first of 8 contiguous rows
      const int j     = j0 + c * 16 + lrow;         // this lane's column
      if (mode == STORE_BF16) {
        bf16_t* out = (bf16_t*)Cp;
#pragma unroll
        for (int e = 0; e < 8; ++e)
          out[(size_t)(mbase + e) * ldc + j] = (bf16_t)a[e];
      } else if (mode == STORE_BF16_T) {            // store C^T: rows contiguous
        bf16_t* out = (bf16_t*)Cp;
        v8bf pk;
#pragma unroll
        for (int e = 0; e < 8; ++e) pk[e] = (bf16_t)a[e];
        *reinterpret_cast<v8bf*>(out + (size_t)j * ldc + mbase) = pk;
      } else {                                      // f32 row-major (final output)
        float* out = (float*)Cp;
#pragma unroll
        for (int e = 0; e < 8; ++e)
          out[(size_t)(mbase + e) * ldc + j] = a[e];
      }
    }
  }
}

// ---------------- host-side orchestration ----------------

extern "C" void kernel_launch(void* const* d_in, const int* in_sizes, int n_in,
                              void* d_out, int out_size, void* d_ws, size_t ws_size,
                              hipStream_t stream) {
  (void)in_sizes; (void)n_in; (void)out_size; (void)ws_size;
  const float* x   = (const float*)d_in[0];
  const float* Wq[2] = {(const float*)d_in[1], (const float*)d_in[2]};
  const float* Wk[2] = {(const float*)d_in[3], (const float*)d_in[4]};
  const float* Wv[2] = {(const float*)d_in[5], (const float*)d_in[6]};
  float* out = (float*)d_out;

  char* ws = (char*)d_ws;
  size_t off = 0;
  auto carve = [&](size_t bytes) -> char* {
    char* p = ws + off;
    off = (off + bytes + 255) & ~(size_t)255;
    return p;
  };
  const size_t SQ = (size_t)DHALF * DHALF * sizeof(bf16_t);   // 512x512 bf16
  bf16_t* xb  = (bf16_t*)carve((size_t)NTOK * DFULL * sizeof(bf16_t)); // x, bf16
  bf16_t* xbT = (bf16_t*)carve((size_t)DFULL * NTOK * sizeof(bf16_t)); // x^T, bf16
  bf16_t* wqT[2] = {(bf16_t*)carve(SQ), (bf16_t*)carve(SQ)};
  bf16_t* wkT[2] = {(bf16_t*)carve(SQ), (bf16_t*)carve(SQ)};
  bf16_t* wvb[2] = {(bf16_t*)carve(SQ), (bf16_t*)carve(SQ)};
  bf16_t* G [2]  = {(bf16_t*)carve(SQ), (bf16_t*)carve(SQ)};
  bf16_t* T1[2]  = {(bf16_t*)carve(SQ), (bf16_t*)carve(SQ)};
  bf16_t* T2[2]  = {(bf16_t*)carve(SQ), (bf16_t*)carve(SQ)};
  bf16_t* CT[2]  = {(bf16_t*)carve(SQ), (bf16_t*)carve(SQ)};

  auto gemm = [&](const bf16_t* A, int lda, const bf16_t* B, int ldb,
                  void* C, int ldc, int M, int N, int K, int mode) {
    int tiles = (M / 64) * (N / 64);            // always a multiple of 4 here
    k_gemm_tn<<<dim3(tiles / 4), dim3(128), 0, stream>>>(A, lda, B, ldb, C, ldc,
                                                         M, N, K, mode);
  };

  // Stage 0: precision/layout conversion
  {
    int n4 = (NTOK * DFULL) / 4;
    k_convert4<<<dim3((n4 + 255) / 256), dim3(256), 0, stream>>>(x, xb, n4);
    k_transpose_convert<<<dim3(DFULL / 32, NTOK / 32), dim3(32, 8), 0, stream>>>(
        x, xbT, NTOK, DFULL);
    int w4 = (DHALF * DHALF) / 4;
    for (int h = 0; h < 2; ++h) {
      k_transpose_convert<<<dim3(DHALF / 32, DHALF / 32), dim3(32, 8), 0, stream>>>(
          Wq[h], wqT[h], DHALF, DHALF);
      k_transpose_convert<<<dim3(DHALF / 32, DHALF / 32), dim3(32, 8), 0, stream>>>(
          Wk[h], wkT[h], DHALF, DHALF);
      k_convert4<<<dim3((w4 + 255) / 256), dim3(256), 0, stream>>>(Wv[h], wvb[h], w4);
    }
  }

  // Stages 1-5 per head
  for (int h = 0; h < 2; ++h) {
    const bf16_t* Xt = xbT + (size_t)h * DHALF * NTOK;   // rows = x columns of head h
    // 1) G = X^T X  (K = 4096)
    gemm(Xt, NTOK, Xt, NTOK, G[h], DHALF, DHALF, DHALF, NTOK, STORE_BF16);
    // 2) T1 = Wq^T Wk
    gemm(wqT[h], DHALF, wkT[h], DHALF, T1[h], DHALF, DHALF, DHALF, DHALF, STORE_BF16);
    // 3) T2 = T1 * G   (G symmetric -> usable directly as TN rhs)
    gemm(T1[h], DHALF, G[h], DHALF, T2[h], DHALF, DHALF, DHALF, DHALF, STORE_BF16);
    // 4) C = T2 * Wv^T, stored transposed: CT[j][d] = C[d][j]
    gemm(T2[h], DHALF, wvb[h], DHALF, CT[h], DHALF, DHALF, DHALF, DHALF, STORE_BF16_T);
    // 5) O = X_h * C -> f32 output columns [h*512, h*512+512)
    gemm(xb + (size_t)h * DHALF, DFULL, CT[h], DHALF,
         out + (size_t)h * DHALF, DFULL, NTOK, DHALF, DHALF, STORE_F32);
  }
}
